// Transformer_76699525972246
// MI455X (gfx1250) — matrix operand order repaired and checked by reading, not compiled, not run on hardware
//
#include <hip/hip_runtime.h>

// ---------------------------------------------------------------------------
// Problem constants (match reference)
// ---------------------------------------------------------------------------
static constexpr int  kB  = 2;
static constexpr int  kS  = 1024;
static constexpr int  kE  = 1024;
static constexpr int  kH  = 16;
static constexpr int  kHD = 64;
static constexpr int  kF  = 4096;
static constexpr int  kV  = 32000;
static constexpr int  kNB = 4;
static constexpr long kBS = (long)kB * kS;   // 2048 rows

// ---------------------------------------------------------------------------
// CDNA5 WMMA types
// ---------------------------------------------------------------------------
typedef __attribute__((ext_vector_type(16))) __bf16   v16bf;
typedef __attribute__((ext_vector_type(8)))  float    v8f;
typedef __attribute__((ext_vector_type(4)))  unsigned u32x4;

union FragAB { v16bf v; unsigned u[8]; };

__device__ __forceinline__ unsigned short f2bf(float f) {
    unsigned u = __float_as_uint(f);
    unsigned r = (u + 0x7FFFu + ((u >> 16) & 1u)) >> 16;   // round-to-nearest-even
    return (unsigned short)r;
}

// ---------------------------------------------------------------------------
// Generic bf16 WMMA GEMM:  C[M,N] = A[M,K] * B + bias   (all dims mult of tile)
//   NTILE : block tile width (64 or 128); block tile is 128 x NTILE, K-step 32
//   TRANSB=true : B given as [N,K] row-major (weight [out,in], or K-matrix)
//   TRANSB=false: B given as [K,N] row-major (weight [in,out], or V-matrix)
//   WF32=true   : write fp32 to Cf ; else write bf16 to Cb
// Batched over blockIdx.z with (outer, inner) strides: z -> zo=z/nInner, zi=z%nInner
// 8 waves (256 thr) as 4(M) x 2(N); wave tile 32 x NTILE/2 (2 x NTILE/32 WMMAs).
// ---------------------------------------------------------------------------
struct GemmP {
    const unsigned short* A;
    const unsigned short* Bm;
    float*          Cf;
    unsigned short* Cb;
    const float*    bias;
    long lda, ldb, ldc;
    long aOut, aIn, bOut, bIn, cOut, cIn;
    int  M, N, K, nInner;
};

template<int NTILE, bool TRANSB, bool RELU, bool WF32>
__global__ __launch_bounds__(256) void gemm_bf16(GemmP p) {
    constexpr int TN = NTILE / 32;          // WMMA n-tiles per wave (2 or 4)
    __shared__ __align__(16) unsigned short sA[128 * 32];
    __shared__ __align__(16) unsigned short sB[NTILE * 32];

    const int tid  = threadIdx.x;
    const int lane = tid & 31;
    const int w    = tid >> 5;
    const int wm   = w & 3;          // wave row   (4 x 32 = 128)
    const int wn   = w >> 2;         // wave col   (2 x NTILE/2)
    const int fr   = lane & 15;      // row/col within 16
    const int kh   = lane >> 4;      // k-half selector

    const long mBase = (long)blockIdx.y * 128;
    const long nBase = (long)blockIdx.x * NTILE;

    const int  z  = (int)blockIdx.z;
    const int  zo = z / p.nInner;
    const int  zi = z - zo * p.nInner;
    const unsigned short* Ab = p.A  + (long)zo * p.aOut + (long)zi * p.aIn;
    const unsigned short* Bb = p.Bm + (long)zo * p.bOut + (long)zi * p.bIn;
    const long cOff = (long)zo * p.cOut + (long)zi * p.cIn;

    v8f acc[2][TN] = {};

    for (int k0 = 0; k0 < p.K; k0 += 32) {
        // ---- stage A tile: 128 x 32 bf16, row-major in LDS -----------------
        #pragma unroll
        for (int pass = 0; pass < 2; ++pass) {
            int unit = tid + pass * 256;           // 0..511, 8 bf16 each
            int row  = unit >> 2;
            int qq   = unit & 3;
            const unsigned short* g = Ab + (mBase + row) * p.lda + k0 + qq * 8;
            *(u32x4*)&sA[row * 32 + qq * 8] = *(const u32x4*)g;
        }
        // ---- stage B tile into sB[n][k] (NTILE x 32) -----------------------
        if (TRANSB) {
            #pragma unroll
            for (int pass = 0; pass < NTILE / 64; ++pass) {
                int unit = tid + pass * 256;       // NTILE*4 units of 8 bf16
                int n    = unit >> 2;
                int qq   = unit & 3;
                const unsigned short* g = Bb + (nBase + n) * p.ldb + k0 + qq * 8;
                *(u32x4*)&sB[n * 32 + qq * 8] = *(const u32x4*)g;
            }
        } else {
            #pragma unroll
            for (int j = 0; j < NTILE / 8; ++j) {
                int e  = tid + j * 256;            // 0..NTILE*32-1
                int kk = e / NTILE;
                int n  = e % NTILE;
                sB[n * 32 + kk] = Bb[(long)(k0 + kk) * p.ldb + nBase + n];
            }
        }
        __syncthreads();

        // ---- gather WMMA fragments per ISA VGPR layouts --------------------
        FragAB a[2], b[TN];
        #pragma unroll
        for (int t = 0; t < 2; ++t) {
            const unsigned short* pa = &sA[(wm * 32 + t * 16 + fr) * 32];
            #pragma unroll
            for (int g2 = 0; g2 < 4; ++g2) {
                a[t].u[g2]     = *(const unsigned*)&pa[kh * 8 + 2 * g2];
                a[t].u[4 + g2] = *(const unsigned*)&pa[16 + kh * 8 + 2 * g2];
            }
        }
        #pragma unroll
        for (int t = 0; t < TN; ++t) {
            const unsigned short* pb = &sB[(wn * (NTILE / 2) + t * 16 + fr) * 32];
            #pragma unroll
            for (int g2 = 0; g2 < 8; ++g2)
                b[t].u[g2] = *(const unsigned*)&pb[(g2 >> 2) * 16 + kh * 8 + 2 * (g2 & 3)];
        }
#if defined(__AMDGCN__)
        #pragma unroll
        for (int tm = 0; tm < 2; ++tm)
            #pragma unroll
            for (int tn = 0; tn < TN; ++tn)
                acc[tm][tn] = __builtin_amdgcn_wmma_f32_16x16x32_bf16(
                    false, a[tm].v, false, b[tn].v,
                    (short)0, acc[tm][tn], false, false);
#endif
        __syncthreads();
    }

    // ---- epilogue: C/D layout VGPR r -> m = r + 8*(lane>=16), n = lane%16 --
    const int rbase = (lane >> 4) * 8;
    #pragma unroll
    for (int tm = 0; tm < 2; ++tm) {
        #pragma unroll
        for (int tn = 0; tn < TN; ++tn) {
            const long gn = nBase + wn * (NTILE / 2) + tn * 16 + fr;
            const float bv = p.bias ? p.bias[gn] : 0.0f;
            #pragma unroll
            for (int r = 0; r < 8; ++r) {
                const long gm = mBase + wm * 32 + tm * 16 + rbase + r;
                float val = acc[tm][tn][r] + bv;
                if (RELU) val = fmaxf(val, 0.0f);
                const long idx = cOff + gm * p.ldc + gn;
                if (WF32) p.Cf[idx] = val;
                else      p.Cb[idx] = f2bf(val);
            }
        }
    }
}

// ---------------------------------------------------------------------------
// fp32 -> bf16 conversion (weights)
// ---------------------------------------------------------------------------
__global__ __launch_bounds__(256) void convert_f32_bf16(const float* in,
                                                        unsigned short* out, long n) {
    long i = (long)blockIdx.x * 256 + threadIdx.x;
    const long stride = (long)gridDim.x * 256;
    for (; i < n; i += stride) out[i] = f2bf(in[i]);
}

// ---------------------------------------------------------------------------
// Embedding: h[b,s,:] = word_embed[x[b,s]] + pos_embed[s]; also bf16 copy
// ---------------------------------------------------------------------------
__global__ __launch_bounds__(256) void embed_kernel(const int* x, const float* we,
                                                    const float* pe, float* h,
                                                    unsigned short* hbf) {
    const long row = blockIdx.x;
    const int  s   = (int)(row % kS);
    const long tok = x[row];
    #pragma unroll
    for (int j = 0; j < 4; ++j) {
        int c = threadIdx.x + j * 256;
        float v = we[tok * kE + c] + pe[(long)s * kE + c];
        h[row * kE + c]   = v;
        hbf[row * kE + c] = f2bf(v);
    }
}

// ---------------------------------------------------------------------------
// Causal softmax row kernel: probs = softmax(scale*scores masked), bf16 out
// one block per (b,h,q) row of length S=1024; masked entries written as 0
// ---------------------------------------------------------------------------
__global__ __launch_bounds__(256) void softmax_causal(const float* sc,
                                                      unsigned short* pr,
                                                      float scale) {
    __shared__ float red[256];
    const int  tid  = threadIdx.x;
    const long row  = blockIdx.x;
    const int  q    = (int)(row & (kS - 1));
    const long base = row * kS;

    float loc[4];
    float mx = -3.0e38f;
    #pragma unroll
    for (int j = 0; j < 4; ++j) {
        int idx = tid + j * 256;
        float v = (idx <= q) ? sc[base + idx] * scale : -3.0e38f;
        loc[j] = v;
        mx = fmaxf(mx, v);
    }
    red[tid] = mx; __syncthreads();
    for (int st = 128; st > 0; st >>= 1) {
        if (tid < st) red[tid] = fmaxf(red[tid], red[tid + st]);
        __syncthreads();
    }
    mx = red[0]; __syncthreads();

    float sum = 0.0f;
    #pragma unroll
    for (int j = 0; j < 4; ++j) {
        int idx = tid + j * 256;
        float e = (idx <= q) ? __expf(loc[j] - mx) : 0.0f;
        loc[j] = e;
        sum += e;
    }
    red[tid] = sum; __syncthreads();
    for (int st = 128; st > 0; st >>= 1) {
        if (tid < st) red[tid] += red[tid + st];
        __syncthreads();
    }
    const float inv = 1.0f / red[0];
    #pragma unroll
    for (int j = 0; j < 4; ++j) {
        int idx = tid + j * 256;
        pr[base + idx] = f2bf(loc[j] * inv);
    }
}

// ---------------------------------------------------------------------------
// h = LayerNorm(src)*g + beta + h  (residual), refresh bf16 copy of h
// ---------------------------------------------------------------------------
__global__ __launch_bounds__(256) void ln_residual_update(const float* src, float* h,
                                                          unsigned short* hbf,
                                                          const float* g,
                                                          const float* beta) {
    __shared__ float red[256];
    const int  tid  = threadIdx.x;
    const long base = (long)blockIdx.x * kE;

    float x[4];
    float s = 0.0f;
    #pragma unroll
    for (int j = 0; j < 4; ++j) {
        x[j] = src[base + tid + j * 256];
        s += x[j];
    }
    red[tid] = s; __syncthreads();
    for (int st = 128; st > 0; st >>= 1) {
        if (tid < st) red[tid] += red[tid + st];
        __syncthreads();
    }
    const float mean = red[0] * (1.0f / kE); __syncthreads();

    float vs = 0.0f;
    #pragma unroll
    for (int j = 0; j < 4; ++j) { float d = x[j] - mean; vs += d * d; }
    red[tid] = vs; __syncthreads();
    for (int st = 128; st > 0; st >>= 1) {
        if (tid < st) red[tid] += red[tid + st];
        __syncthreads();
    }
    const float rstd = rsqrtf(red[0] * (1.0f / kE) + 1e-6f);

    #pragma unroll
    for (int j = 0; j < 4; ++j) {
        int c = tid + j * 256;
        float out = (x[j] - mean) * rstd * g[c] + beta[c] + h[base + c];
        h[base + c]   = out;
        hbf[base + c] = f2bf(out);
    }
}

// ---------------------------------------------------------------------------
// Host-side orchestration
// ---------------------------------------------------------------------------
extern "C" void kernel_launch(void* const* d_in, const int* in_sizes, int n_in,
                              void* d_out, int out_size, void* d_ws, size_t ws_size,
                              hipStream_t stream) {
    (void)in_sizes; (void)n_in; (void)out_size; (void)ws_size;

    const int*   xin = (const int*)  d_in[0];
    const float* we  = (const float*)d_in[1];
    const float* pe  = (const float*)d_in[2];
    const float* WK  = (const float*)d_in[3];
    const float* WQ  = (const float*)d_in[4];
    const float* WV  = (const float*)d_in[5];
    const float* WO  = (const float*)d_in[6];
    const float* Wup = (const float*)d_in[7];
    const float* bup = (const float*)d_in[8];
    const float* Wdn = (const float*)d_in[9];
    const float* bdn = (const float*)d_in[10];
    const float* g1  = (const float*)d_in[11];
    const float* be1 = (const float*)d_in[12];
    const float* g2  = (const float*)d_in[13];
    const float* be2 = (const float*)d_in[14];
    const float* Wun = (const float*)d_in[15];
    const float* bun = (const float*)d_in[16];
    float* out = (float*)d_out;

    // --- workspace carve-out (256B aligned) ---
    char* wsb = (char*)d_ws;
    size_t cur = 0;
    auto alloc = [&](size_t bytes) -> void* {
        cur = (cur + 255) & ~(size_t)255;
        void* p = wsb + cur;
        cur += bytes;
        return p;
    };
    const size_t nQKV = (size_t)kNB * kE * kE;       // per WK/WQ/WV/WO tensor
    const size_t nUP  = (size_t)kNB * kE * kF;
    const size_t nUN  = (size_t)kE * kV;

    unsigned short* wkb  = (unsigned short*)alloc(nQKV * 2);
    unsigned short* wqb  = (unsigned short*)alloc(nQKV * 2);
    unsigned short* wvb  = (unsigned short*)alloc(nQKV * 2);
    unsigned short* wob  = (unsigned short*)alloc(nQKV * 2);
    unsigned short* wupb = (unsigned short*)alloc(nUP * 2);
    unsigned short* wdnb = (unsigned short*)alloc(nUP * 2);
    unsigned short* wunb = (unsigned short*)alloc(nUN * 2);

    float*          h32   = (float*)         alloc((size_t)kBS * kE * 4);
    unsigned short* hbf   = (unsigned short*)alloc((size_t)kBS * kE * 2);
    unsigned short* qb    = (unsigned short*)alloc((size_t)kBS * kE * 2);
    unsigned short* kb    = (unsigned short*)alloc((size_t)kBS * kE * 2);
    unsigned short* vb    = (unsigned short*)alloc((size_t)kBS * kE * 2);
    unsigned short* attnb = (unsigned short*)alloc((size_t)kBS * kE * 2);
    float*          o32   = (float*)         alloc((size_t)kBS * kE * 4);
    unsigned short* fb    = (unsigned short*)alloc((size_t)kBS * kF * 2);
    float*          scores= (float*)         alloc((size_t)kB * kH * kS * kS * 4);
    unsigned short* probs = (unsigned short*)alloc((size_t)kB * kH * kS * kS * 2);

    // --- one-time weight conversion to bf16 ---
    convert_f32_bf16<<<2048, 256, 0, stream>>>(WK,  wkb,  (long)nQKV);
    convert_f32_bf16<<<2048, 256, 0, stream>>>(WQ,  wqb,  (long)nQKV);
    convert_f32_bf16<<<2048, 256, 0, stream>>>(WV,  wvb,  (long)nQKV);
    convert_f32_bf16<<<2048, 256, 0, stream>>>(WO,  wob,  (long)nQKV);
    convert_f32_bf16<<<2048, 256, 0, stream>>>(Wup, wupb, (long)nUP);
    convert_f32_bf16<<<2048, 256, 0, stream>>>(Wdn, wdnb, (long)nUP);
    convert_f32_bf16<<<2048, 256, 0, stream>>>(Wun, wunb, (long)nUN);

    // --- embedding ---
    embed_kernel<<<(int)kBS, 256, 0, stream>>>(xin, we, pe, h32, hbf);

    for (int i = 0; i < kNB; ++i) {
        const unsigned short* wq_i = wqb + (size_t)i * kE * kE;
        const unsigned short* wk_i = wkb + (size_t)i * kE * kE;
        const unsigned short* wv_i = wvb + (size_t)i * kE * kE;
        const unsigned short* wo_i = wob + (size_t)i * kE * kE;

        // --- QKV projections: [BS,E] x [E,E]^T -> bf16 [B,S,E] ---
        GemmP p; p.lda = kE; p.ldb = kE; p.ldc = kE;
        p.aOut = p.aIn = p.bOut = p.bIn = p.cOut = p.cIn = 0;
        p.A = hbf; p.Cf = nullptr; p.bias = nullptr;
        p.M = (int)kBS; p.N = kE; p.K = kE; p.nInner = 1;
        dim3 gProj(kE / 128, (int)(kBS / 128), 1);
        p.Bm = wq_i; p.Cb = qb; gemm_bf16<128, true,  false, false><<<gProj, 256, 0, stream>>>(p);
        p.Bm = wk_i; p.Cb = kb; gemm_bf16<128, true,  false, false><<<gProj, 256, 0, stream>>>(p);
        p.Bm = wv_i; p.Cb = vb; gemm_bf16<128, true,  false, false><<<gProj, 256, 0, stream>>>(p);

        // --- scores = Q K^T per (b,h): [S,HD] x [S,HD]^T -> fp32 [B*H,S,S] ---
        p.A = qb;  p.lda = kE; p.aOut = (long)kS * kE;     p.aIn = kHD;
        p.Bm = kb; p.ldb = kE; p.bOut = (long)kS * kE;     p.bIn = kHD;
        p.Cf = scores; p.Cb = nullptr; p.ldc = kS;
        p.cOut = (long)kH * kS * kS; p.cIn = (long)kS * kS;
        p.bias = nullptr; p.M = kS; p.N = kS; p.K = kHD; p.nInner = kH;
        gemm_bf16<128, true, false, true><<<dim3(kS / 128, kS / 128, kB * kH), 256, 0, stream>>>(p);

        // --- causal softmax (scale = 1/sqrt(HD)) ---
        softmax_causal<<<kB * kH * kS, 256, 0, stream>>>(scores, probs, 0.125f);

        // --- O = P V per (b,h): [S,S] x [S,HD] -> bf16 written into [B,S,E] ---
        p.A = probs; p.lda = kS; p.aOut = (long)kH * kS * kS; p.aIn = (long)kS * kS;
        p.Bm = vb;   p.ldb = kE; p.bOut = (long)kS * kE;      p.bIn = kHD;
        p.Cf = nullptr; p.Cb = attnb; p.ldc = kE;
        p.cOut = (long)kS * kE; p.cIn = kHD;
        p.bias = nullptr; p.M = kS; p.N = kHD; p.K = kS; p.nInner = kH;
        gemm_bf16<64, false, false, false><<<dim3(kHD / 64, kS / 128, kB * kH), 256, 0, stream>>>(p);

        // --- attn @ WO : [BS,E] x [E,E] -> fp32 ---
        p.A = attnb; p.lda = kE; p.aOut = p.aIn = 0;
        p.Bm = wo_i; p.ldb = kE; p.bOut = p.bIn = 0;
        p.Cf = o32; p.Cb = nullptr; p.ldc = kE; p.cOut = p.cIn = 0;
        p.bias = nullptr; p.M = (int)kBS; p.N = kE; p.K = kE; p.nInner = 1;
        gemm_bf16<128, false, false, true><<<gProj, 256, 0, stream>>>(p);

        // --- h = LN(o)*g1+b1 + h ---
        ln_residual_update<<<(int)kBS, 256, 0, stream>>>(o32, h32, hbf,
                                                         g1 + (size_t)i * kE,
                                                         be1 + (size_t)i * kE);

        // --- FFN up: relu(h @ W_up + b_up) -> bf16 [BS,F] ---
        p.A = hbf; p.lda = kE;
        p.Bm = wupb + (size_t)i * kE * kF; p.ldb = kF;
        p.Cf = nullptr; p.Cb = fb; p.ldc = kF;
        p.bias = bup + (size_t)i * kF;
        p.M = (int)kBS; p.N = kF; p.K = kE; p.nInner = 1;
        gemm_bf16<128, false, true, false><<<dim3(kF / 128, (int)(kBS / 128), 1), 256, 0, stream>>>(p);

        // --- FFN down: f @ W_down + b_down -> fp32 ---
        p.A = fb; p.lda = kF;
        p.Bm = wdnb + (size_t)i * kF * kE; p.ldb = kE;
        p.Cf = o32; p.Cb = nullptr; p.ldc = kE;
        p.bias = bdn + (size_t)i * kE;
        p.M = (int)kBS; p.N = kE; p.K = kF; p.nInner = 1;
        gemm_bf16<128, false, false, true><<<gProj, 256, 0, stream>>>(p);

        // --- h = LN(f)*g2+b2 + h ---
        ln_residual_update<<<(int)kBS, 256, 0, stream>>>(o32, h32, hbf,
                                                         g2 + (size_t)i * kE,
                                                         be2 + (size_t)i * kE);
    }

    // --- unembed: logits = h @ W_un + b_un -> fp32 d_out [BS, V] ---
    GemmP p; p.A = hbf; p.lda = kE; p.aOut = p.aIn = 0;
    p.Bm = wunb; p.ldb = kV; p.bOut = p.bIn = 0;
    p.Cf = out; p.Cb = nullptr; p.ldc = kV; p.cOut = p.cIn = 0;
    p.bias = bun; p.M = (int)kBS; p.N = kV; p.K = kE; p.nInner = 1;
    gemm_bf16<128, false, false, true><<<dim3(kV / 128, (int)(kBS / 128), 1), 256, 0, stream>>>(p);
}
